// GenActivation_5909874999451
// MI455X (gfx1250) — matrix operand order repaired and checked by reading, not compiled, not run on hardware
//
#include <hip/hip_runtime.h>

// Problem constants (match reference)
constexpr int Bb = 8192;   // batch
constexpr int Dd = 1024;   // input dim (K)
constexpr int Nn = 4096;   // outputs
constexpr float TAU_INV = 5.0f;  // 1/0.2

#define BM 128
#define BN 128
#define BK 32
#define LDK 40              // padded K stride in bf16 elements (80B, multiple of 16B)
#define TILE (BM * LDK)     // one LDS buffer in elements

typedef __attribute__((ext_vector_type(16))) __bf16 v16bf;
typedef __attribute__((ext_vector_type(8)))  __bf16 bf8;
typedef __attribute__((ext_vector_type(4)))  __bf16 bf4;
typedef __attribute__((ext_vector_type(2)))  __bf16 bf2;
typedef __attribute__((ext_vector_type(8)))  float  v8f;

// ds_swizzle group-of-32 xor pattern: and_mask=0x1f, or=0, xor=m (ISA §11)
#define SWZ_XOR(v, m) \
  __int_as_float(__builtin_amdgcn_ds_swizzle(__float_as_int(v), (((m) << 10) | 0x1f)))

__global__ __launch_bounds__(256)
void gen_act_gemm(const float* __restrict__ x, const float* __restrict__ W,
                  const float* __restrict__ bias, const float* __restrict__ gumbel,
                  const int* __restrict__ flags,
                  float* __restrict__ out, float* __restrict__ act)
{
  __shared__ __bf16 aS[2 * TILE];   // x tile, row-major [row][k], ping-pong
  __shared__ __bf16 wS[2 * TILE];   // W tile, col-major [col][k], ping-pong

  const int tid   = threadIdx.x;
  const int wid   = tid >> 5;      // 0..7 (8 waves of 32)
  const int lane  = tid & 31;
  const int half  = lane >> 4;     // lane half (0/1)
  const int l16   = lane & 15;
  const int waveM = wid >> 2;      // 0..1  -> 64 rows each
  const int waveN = wid & 3;       // 0..3  -> 32 cols each

  const int m0 = blockIdx.y * BM;
  const int n0 = blockIdx.x * BN;

  v8f acc[4][2];
  #pragma unroll
  for (int i = 0; i < 4; ++i)
    #pragma unroll
    for (int j = 0; j < 2; ++j)
      acc[i][j] = {};

  // x tile (128 rows x 32 k): thread -> 4x float4 (rows xr+32i, cols xc..xc+3)
  const int xr = tid >> 3;          // 0..31 base row
  const int xc = (tid & 7) * 4;     // k offset 0,4,...,28
  // W tile (32 k x 128 cols): thread owns 2 consecutive k-rows x 8 cols
  const int wk = (tid >> 4) * 2;    // k-row pair base: 0,2,...,30
  const int wc = (tid & 15) * 8;    // col offset 0..120

  float4 xg[4], wg[4];
  auto loadg = [&](int k0) {
    #pragma unroll
    for (int i = 0; i < 4; ++i)
      xg[i] = *(const float4*)(x + (size_t)(m0 + xr + 32 * i) * Dd + k0 + xc);
    #pragma unroll
    for (int r = 0; r < 2; ++r)
      #pragma unroll
      for (int j = 0; j < 2; ++j)
        wg[2*r + j] = *(const float4*)(W + (size_t)(k0 + wk + r) * Nn + n0 + wc + 4 * j);
  };

  loadg(0);  // prime the pipeline

  constexpr int KT = Dd / BK;      // 32 K-steps
  for (int kt = 0; kt < KT; ++kt) {
    const int buf = (kt & 1) * TILE;

    // drain staged registers into LDS (native f32->bf16 RNE casts)
    #pragma unroll
    for (int i = 0; i < 4; ++i) {
      bf4 h;
      h.x = (__bf16)xg[i].x; h.y = (__bf16)xg[i].y;
      h.z = (__bf16)xg[i].z; h.w = (__bf16)xg[i].w;
      *(bf4*)(&aS[buf + (xr + 32 * i) * LDK + xc]) = h;
    }
    // transposed W store: per column, pack (k, k+1) pair -> one b32 store
    {
      const float* r0 = (const float*)&wg[0];   // cols wc..wc+7, k-row wk
      const float* r1 = (const float*)&wg[2];   // cols wc..wc+7, k-row wk+1
      #pragma unroll
      for (int c = 0; c < 8; ++c) {
        bf2 p; p.x = (__bf16)r0[c]; p.y = (__bf16)r1[c];
        *(bf2*)(&wS[buf + (wc + c) * LDK + wk]) = p;
      }
    }
    __syncthreads();

    // issue next K-step's global loads; they overlap the WMMAs below
    if (kt + 1 < KT) loadg((kt + 1) * BK);

    // A fragment (16x32 bf16): lanes 0-15 hold K 0-7 / 16-23; lanes 16-31 K 8-15 / 24-31
    union FA { bf8 h[2]; v16bf v; } a[4];
    const int kA0 = 8 * half;
    const int kA1 = 16 + 8 * half;
    #pragma unroll
    for (int mi = 0; mi < 4; ++mi) {
      const int row = waveM * 64 + mi * 16 + l16;
      a[mi].h[0] = *(const bf8*)(&aS[buf + row * LDK + kA0]);
      a[mi].h[1] = *(const bf8*)(&aS[buf + row * LDK + kA1]);
    }
    // B fragment (32x16 bf16): N = lane%16; lanes 0-15 K 0-7 / 16-23; lanes 16-31 K 8-15 / 24-31
    union FB { bf8 h[2]; v16bf v; } bfr[2];
    const int kB = 8 * half;
    #pragma unroll
    for (int ni = 0; ni < 2; ++ni) {
      const int col = waveN * 32 + ni * 16 + l16;
      bfr[ni].h[0] = *(const bf8*)(&wS[buf + col * LDK + kB]);
      bfr[ni].h[1] = *(const bf8*)(&wS[buf + col * LDK + kB + 16]);
    }

    #pragma unroll
    for (int mi = 0; mi < 4; ++mi)
      #pragma unroll
      for (int ni = 0; ni < 2; ++ni)
        acc[mi][ni] = __builtin_amdgcn_wmma_f32_16x16x32_bf16(
            false, a[mi].v, false, bfr[ni].v, (short)0, acc[mi][ni], false, false);
    // no trailing barrier needed: each wave's dscnt-wait before its WMMAs orders
    // its buffer reads before its next barrier signal (ping-pong WAR is 2 steps away)
  }

  // Fused epilogue. C layout: VGPR r -> row (r + 8*half), lane%16 -> col.
  // Segment width 16 == WMMA N-tile -> softmax reduces over the 16-lane group
  // with single-instruction ds_swizzle xor butterflies.
  #pragma unroll
  for (int ni = 0; ni < 2; ++ni) {
    const int colBase = n0 + waveN * 32 + ni * 16;
    const int col  = colBase + l16;
    const int flag = flags[colBase >> 4];       // wave-uniform per N-tile
    const float bv = bias[col];
    #pragma unroll
    for (int mi = 0; mi < 4; ++mi) {
      #pragma unroll
      for (int r = 0; r < 8; ++r) {
        const int row = m0 + waveM * 64 + mi * 16 + r + 8 * half;
        const size_t idx = (size_t)row * Nn + col;
        const float y = acc[mi][ni][r] + bv;
        out[idx] = y;
        float res;
        if (flag == 0) {
          res = tanhf(y);
        } else {
          const float t = (y + gumbel[idx]) * TAU_INV;
          float m = t;
          m = fmaxf(m, SWZ_XOR(m, 1));
          m = fmaxf(m, SWZ_XOR(m, 2));
          m = fmaxf(m, SWZ_XOR(m, 4));
          m = fmaxf(m, SWZ_XOR(m, 8));
          const float e = __expf(t - m);
          float s = e;
          s += SWZ_XOR(s, 1);
          s += SWZ_XOR(s, 2);
          s += SWZ_XOR(s, 4);
          s += SWZ_XOR(s, 8);
          res = e / s;
        }
        act[idx] = res;
      }
    }
  }
}

extern "C" void kernel_launch(void* const* d_in, const int* in_sizes, int n_in,
                              void* d_out, int out_size, void* d_ws, size_t ws_size,
                              hipStream_t stream) {
  const float* x      = (const float*)d_in[0];
  const float* W      = (const float*)d_in[1];
  const float* b      = (const float*)d_in[2];
  const float* gumbel = (const float*)d_in[3];
  const int*   flags  = (const int*)d_in[4];

  float* out = (float*)d_out;                      // outputs [B,N]
  float* act = out + (size_t)Bb * Nn;              // data_t  [B,N]

  dim3 grid(Nn / BN, Bb / BM);                     // 32 x 64 blocks
  gen_act_gemm<<<grid, 256, 0, stream>>>(x, W, b, gumbel, flags, out, act);
}
